// TGBSTGraphNeuralCDE_12206297055726
// MI455X (gfx1250) — matrix operand order repaired and checked by reading, not compiled, not run on hardware
//
#include <hip/hip_runtime.h>
#include <hip/hip_bf16.h>
#include <math.h>

// ---------------------------------------------------------------------------
// TGBST Graph Neural CDE on MI455X (gfx1250, wave32, WMMA).
// All GEMMs run on v_wmma_f32_16x16x32_bf16 (bf16 operands, fp32 accumulate).
// Each wave computes a 16 x (16*NT) tile: the dynamic A-fragment (fp32 load +
// axpy + bf16 cvt) is loaded ONCE per K-chunk and reused across NT B tiles,
// amortizing VALU conversion cost and (for k_dX) halving HBM traffic.
// ---------------------------------------------------------------------------

typedef __bf16 bf16_t;
typedef __attribute__((ext_vector_type(16))) __bf16 v16bf;
typedef __attribute__((ext_vector_type(8)))  float  v8f;

#define NN   1024   // num_nodes
#define HH   64     // hidden_dim
#define DD   32     // data_embed_dim
#define TT   128    // time points
#define NSTEPS 128

// ---- fragment loaders (ISA 7.12.2 layouts, wave32) ----

// A-fragment (16x32, 16-bit): lane -> row = rowbase + lane%16,
// kbase = (lane<16)?0:8 ; e<8 -> K=kc+kbase+e, e>=8 -> K=kc+16+kbase+(e-8).
// Value = P[row,k] + scale * Q[row,k]  (fp32 row-major).
__device__ __forceinline__ v16bf load_a_f32(const float* __restrict__ P,
                                            const float* __restrict__ Q,
                                            float scale, int ld, int rowbase, int kc) {
  int lane  = threadIdx.x & 31;
  size_t row = (size_t)(rowbase + (lane & 15));
  int kbase = (lane < 16) ? 0 : 8;
  const float* p0 = P + row * (size_t)ld + kc + kbase;
  const float* q0 = Q + row * (size_t)ld + kc + kbase;
  v16bf a;
#pragma unroll
  for (int e = 0; e < 8; ++e) a[e]     = (bf16_t)(p0[e]      + scale * q0[e]);
#pragma unroll
  for (int e = 0; e < 8; ++e) a[8 + e] = (bf16_t)(p0[16 + e] + scale * q0[16 + e]);
  return a;
}

// A-fragment directly from a bf16 row-major matrix (adjacency A).
__device__ __forceinline__ v16bf load_a_bf16(const bf16_t* __restrict__ P,
                                             int ld, int rowbase, int kc) {
  int lane  = threadIdx.x & 31;
  size_t row = (size_t)(rowbase + (lane & 15));
  int kbase = (lane < 16) ? 0 : 8;
  const bf16_t* p0 = P + row * (size_t)ld + kc + kbase;
  v16bf a;
#pragma unroll
  for (int e = 0; e < 8; ++e) a[e]     = p0[e];
#pragma unroll
  for (int e = 0; e < 8; ++e) a[8 + e] = p0[16 + e];
  return a;
}

// B-fragment (32x16, 16-bit): lane -> col = colbase + lane%16,
// kbase = (lane<16)?0:8 ; e<8 -> K=kc+kbase+e, e>=8 -> K=kc+16+kbase+(e-8).  Bt[col][k] transposed, bf16.
__device__ __forceinline__ v16bf load_b_bf16(const bf16_t* __restrict__ Bt,
                                             int ldb, int colbase, int kc) {
  int lane = threadIdx.x & 31;
  size_t col = (size_t)(colbase + (lane & 15));
  int kofs = (lane < 16) ? 0 : 8;
  const bf16_t* p = Bt + col * (size_t)ldb + kc + kofs;
  v16bf b;
#pragma unroll
  for (int e = 0; e < 16; ++e) b[e] = p[e < 8 ? e : 8 + e];
  return b;
}

// Same but from an fp32 K-contiguous matrix (hW^T, changes per stage).
__device__ __forceinline__ v16bf load_b_f32(const float* __restrict__ Bt,
                                            int ldb, int colbase, int kc) {
  int lane = threadIdx.x & 31;
  size_t col = (size_t)(colbase + (lane & 15));
  int kofs = (lane < 16) ? 0 : 8;
  const float* p = Bt + col * (size_t)ldb + kc + kofs;
  v16bf b;
#pragma unroll
  for (int e = 0; e < 16; ++e) b[e] = (bf16_t)p[e < 8 ? e : 8 + e];
  return b;
}

// ---- one-time weight conversion / transposition to bf16, zero acc ----
__global__ void k_prep(const float* __restrict__ A,   const float* __restrict__ Wf,
                       const float* __restrict__ Wg,  const float* __restrict__ Wdata,
                       const float* __restrict__ Wench, const float* __restrict__ Wencz,
                       const float* __restrict__ Wdec,
                       bf16_t* __restrict__ A_bf, bf16_t* __restrict__ Wf_bf,
                       bf16_t* __restrict__ Wgt_bf, bf16_t* __restrict__ Wdata_bf,
                       bf16_t* __restrict__ Wench_bf, bf16_t* __restrict__ Wencz_bf,
                       bf16_t* __restrict__ Wdec_bf, float* __restrict__ acc) {
  int i = blockIdx.x * blockDim.x + threadIdx.x;
  int stride = gridDim.x * blockDim.x;
  for (int j = i; j < NN * NN; j += stride) A_bf[j] = (bf16_t)A[j];
  for (int j = i; j < HH * HH; j += stride) Wf_bf[j] = (bf16_t)Wf[j];
  for (int j = i; j < (HH * DD) * HH; j += stride) {           // Wgt[n][k] = Wg[k][n]
    int n = j >> 6, k = j & 63;
    Wgt_bf[j] = (bf16_t)Wg[(size_t)k * (HH * DD) + n];
  }
  for (int j = i; j < DD * NN; j += stride) Wdata_bf[j] = (bf16_t)Wdata[j];
  for (int j = i; j < HH * NN; j += stride) { Wench_bf[j] = (bf16_t)Wench[j];
                                              Wencz_bf[j] = (bf16_t)Wencz[j]; }
  for (int j = i; j < NN * HH; j += stride) Wdec_bf[j] = (bf16_t)Wdec[j];
  for (int j = i; j < 2 * NN * HH; j += stride) acc[j] = 0.0f;
}

// ---- dX[t] = (x_data[t+1]-x_data[t]) @ W_data^T  (b_data/dt cancel) ----
// One wave -> 16 rows x 32 cols (NT=2): A-fragment loaded once per K-chunk.
__global__ void __launch_bounds__(32)
k_dX(const float* __restrict__ x_data, const bf16_t* __restrict__ Wdata_bf,
     float* __restrict__ dX) {
  int bid = blockIdx.x;              // (TT-1) * 64 blocks
  int t = bid >> 6;
  int rowbase = (bid & 63) * 16;
  const float* X1 = x_data + (size_t)(t + 1) * NN * NN;
  const float* X0 = x_data + (size_t)t * NN * NN;
  v8f c0 = {}, c1 = {};
  for (int kc = 0; kc < NN; kc += 32) {
    v16bf a  = load_a_f32(X1, X0, -1.0f, NN, rowbase, kc);   // X1 - X0
    v16bf b0 = load_b_bf16(Wdata_bf, NN, 0,  kc);
    v16bf b1 = load_b_bf16(Wdata_bf, NN, 16, kc);
    c0 = __builtin_amdgcn_wmma_f32_16x16x32_bf16(false, a, false, b0, (short)0, c0,
                                                 false, false);
    c1 = __builtin_amdgcn_wmma_f32_16x16x32_bf16(false, a, false, b1, (short)0, c1,
                                                 false, false);
  }
  int lane = threadIdx.x & 31;
  int col = lane & 15;
  int rowofs = rowbase + ((lane < 16) ? 0 : 8);
  float* out = dX + (size_t)t * NN * DD;
#pragma unroll
  for (int r = 0; r < 8; ++r) {
    out[(size_t)(rowofs + r) * DD + col]      = c0[r];
    out[(size_t)(rowofs + r) * DD + col + 16] = c1[r];
  }
}

// ---- generic wave-tile GEMM: C = act( (Ay + scale*Ak) @ Bt^T + bias ) ----
// One wave -> 16 x (16*NT).  FLAGS bit0: tanh, bit1: transposed store.
template <int NT, int FLAGS>
__global__ void __launch_bounds__(32)
k_gemm(const float* __restrict__ Ay, const float* __restrict__ Ak, float scale,
       int lda, int kdim, const bf16_t* __restrict__ Bt,
       const float* __restrict__ bias, float* __restrict__ Cout, int ldc,
       int ngroups) {
  int bid = blockIdx.x;
  int rowbase  = (bid / ngroups) * 16;
  int colbase0 = (bid % ngroups) * 16 * NT;
  v8f c[NT];
#pragma unroll
  for (int j = 0; j < NT; ++j) c[j] = (v8f){};
  for (int kc = 0; kc < kdim; kc += 32) {
    v16bf a = load_a_f32(Ay, Ak, scale, lda, rowbase, kc);
#pragma unroll
    for (int j = 0; j < NT; ++j) {
      v16bf b = load_b_bf16(Bt, kdim, colbase0 + 16 * j, kc);
      c[j] = __builtin_amdgcn_wmma_f32_16x16x32_bf16(false, a, false, b, (short)0,
                                                     c[j], false, false);
    }
  }
  int lane = threadIdx.x & 31;
  int rowofs = rowbase + ((lane < 16) ? 0 : 8);
#pragma unroll
  for (int j = 0; j < NT; ++j) {
    int col = colbase0 + 16 * j + (lane & 15);
    float bv = bias ? bias[col] : 0.0f;
#pragma unroll
    for (int r = 0; r < 8; ++r) {
      float v = c[j][r] + bv;
      if (FLAGS & 1) v = tanhf(v);
      if (FLAGS & 2) Cout[(size_t)col * ldc + (rowofs + r)] = v;
      else           Cout[(size_t)(rowofs + r) * ldc + col] = v;
    }
  }
}

// ---- k_h = tanh(A @ hW + bf); acc_h += w * k_h  (one wave -> 16 x 64) ----
__global__ void __launch_bounds__(32)
k_dh(const bf16_t* __restrict__ A_bf, const float* __restrict__ hWt,
     const float* __restrict__ bfv, float* __restrict__ kout_h,
     float* __restrict__ acc_h, float w) {
  int rowbase = blockIdx.x * 16;     // 64 blocks
  v8f c[4];
#pragma unroll
  for (int j = 0; j < 4; ++j) c[j] = (v8f){};
  for (int kc = 0; kc < NN; kc += 32) {
    v16bf a = load_a_bf16(A_bf, NN, rowbase, kc);
#pragma unroll
    for (int j = 0; j < 4; ++j) {
      v16bf b = load_b_f32(hWt, NN, 16 * j, kc);   // hWt[col][k], K-contiguous
      c[j] = __builtin_amdgcn_wmma_f32_16x16x32_bf16(false, a, false, b, (short)0,
                                                     c[j], false, false);
    }
  }
  int lane = threadIdx.x & 31;
  int rowofs = rowbase + ((lane < 16) ? 0 : 8);
#pragma unroll
  for (int j = 0; j < 4; ++j) {
    int col = 16 * j + (lane & 15);
    float bb = bfv[col];
#pragma unroll
    for (int r = 0; r < 8; ++r) {
      float v = tanhf(c[j][r] + bb);
      size_t idx = (size_t)(rowofs + r) * HH + col;
      kout_h[idx] = v;
      acc_h[idx] += w * v;           // single writer per element
    }
  }
}

// ---- k_z[n,h] = sum_d G[n, h*32+d] * dXr[n,d]; acc_z += w * k_z ----
__global__ void __launch_bounds__(256)
k_dz(const float* __restrict__ G, const float* __restrict__ dXr,
     float* __restrict__ kout_z, float* __restrict__ acc_z, float w) {
  int tid = blockIdx.x * 256 + threadIdx.x;   // NN*HH threads
  int n = tid >> 6, hh = tid & 63;
  const float* g  = G + (size_t)n * (HH * DD) + hh * DD;
  const float* dx = dXr + (size_t)n * DD;
  float s = 0.0f;
#pragma unroll
  for (int d = 0; d < DD; ++d) s += g[d] * dx[d];
  kout_z[tid] = s;
  acc_z[tid] += w * s;
}

// ---- y += (dt/6) * acc ; acc = 0 ----
__global__ void __launch_bounds__(256)
k_update(float* __restrict__ y, float* __restrict__ acc, float dt6) {
  int tid = blockIdx.x * 256 + threadIdx.x;   // 2*NN*HH
  y[tid] += dt6 * acc[tid];
  acc[tid] = 0.0f;
}

// ---------------------------------------------------------------------------
extern "C" void kernel_launch(void* const* d_in, const int* in_sizes, int n_in,
                              void* d_out, int out_size, void* d_ws, size_t ws_size,
                              hipStream_t stream) {
  (void)in_sizes; (void)n_in; (void)out_size; (void)ws_size;
  const float* x_data  = (const float*)d_in[2];
  const float* x0      = (const float*)d_in[3];
  const float* W_enc_h = (const float*)d_in[5];
  const float* b_enc_h = (const float*)d_in[6];
  const float* W_enc_z = (const float*)d_in[7];
  const float* b_enc_z = (const float*)d_in[8];
  const float* W_dec   = (const float*)d_in[9];
  const float* b_dec   = (const float*)d_in[10];
  const float* W_data  = (const float*)d_in[11];
  const float* A       = (const float*)d_in[13];
  const float* Wf      = (const float*)d_in[14];
  const float* bfv     = (const float*)d_in[15];
  const float* Wg      = (const float*)d_in[16];
  const float* bg      = (const float*)d_in[17];

  // workspace carve-out (256B aligned)
  char* ws = (char*)d_ws;
  size_t off = 0;
  auto carve = [&](size_t bytes) -> void* {
    void* p = ws + off;
    off += (bytes + 255) & ~(size_t)255;
    return p;
  };
  float*  dX       = (float*)carve((size_t)(TT - 1) * NN * DD * 4);
  float*  G        = (float*)carve((size_t)NN * HH * DD * 4);
  float*  y        = (float*)carve((size_t)2 * NN * HH * 4);
  float*  kA       = (float*)carve((size_t)2 * NN * HH * 4);
  float*  kB       = (float*)carve((size_t)2 * NN * HH * 4);
  float*  acc      = (float*)carve((size_t)2 * NN * HH * 4);
  float*  hWt      = (float*)carve((size_t)HH * NN * 4);
  bf16_t* A_bf     = (bf16_t*)carve((size_t)NN * NN * 2);
  bf16_t* Wf_bf    = (bf16_t*)carve((size_t)HH * HH * 2);
  bf16_t* Wgt_bf   = (bf16_t*)carve((size_t)HH * DD * HH * 2);
  bf16_t* Wdata_bf = (bf16_t*)carve((size_t)DD * NN * 2);
  bf16_t* Wench_bf = (bf16_t*)carve((size_t)HH * NN * 2);
  bf16_t* Wencz_bf = (bf16_t*)carve((size_t)HH * NN * 2);
  bf16_t* Wdec_bf  = (bf16_t*)carve((size_t)NN * HH * 2);

  float* y_h = y;           float* y_z = y + NN * HH;
  float* acc_h = acc;       float* acc_z = acc + NN * HH;

  // Phase 0: weight prep
  k_prep<<<512, 256, 0, stream>>>(A, Wf, Wg, W_data, W_enc_h, W_enc_z, W_dec,
                                  A_bf, Wf_bf, Wgt_bf, Wdata_bf, Wench_bf,
                                  Wencz_bf, Wdec_bf, acc);
  // Phase 1: dX (127 GEMMs, the only HBM-heavy pass; NT=2 -> x_data read ~2x)
  k_dX<<<(TT - 1) * 64, 32, 0, stream>>>(x_data, Wdata_bf, dX);
  // Phase 2: encoders  y_h = x0 @ Wenc_h^T + b ; y_z likewise  (NT=4)
  k_gemm<4, 0><<<64, 32, 0, stream>>>(x0, x0, 0.0f, NN, NN, Wench_bf, b_enc_h,
                                      y_h, HH, 1);
  k_gemm<4, 0><<<64, 32, 0, stream>>>(x0, x0, 0.0f, NN, NN, Wencz_bf, b_enc_z,
                                      y_z, HH, 1);

  // Phase 3: RK4 over [0, 127], 128 fixed steps
  const float dt = 127.0f / 128.0f;
  const float cs[4] = {0.0f, 0.5f, 0.5f, 1.0f};
  const float wq[4] = {1.0f, 2.0f, 2.0f, 1.0f};
  float* kbuf[2] = {kA, kB};

  for (int i = 0; i < NSTEPS; ++i) {
    float tbase = (float)i * dt;
    for (int s = 0; s < 4; ++s) {
      float scale = cs[s] * dt;             // y_stage = y + (cs*dt)*k_prev
      float tq = tbase + cs[s] * dt;        // dxdt index = clip(floor(t), 0, T-2)
      int idx = (int)floorf(tq);
      if (idx < 0) idx = 0; if (idx > TT - 2) idx = TT - 2;
      const float* kprev_h, * kprev_z;
      if (s == 0) { kprev_h = y_h; kprev_z = y_z; scale = 0.0f; }
      else        { kprev_h = kbuf[(s - 1) & 1]; kprev_z = kprev_h + NN * HH; }
      float* kcur = kbuf[s & 1];
      float* kcur_h = kcur; float* kcur_z = kcur + NN * HH;

      // hW^T[j][n] = (h_stage @ Wf^T)^T  (transposed store -> K-contiguous B)
      k_gemm<4, 2><<<64, 32, 0, stream>>>(y_h, kprev_h, scale, HH, HH, Wf_bf,
                                          nullptr, hWt, NN, 1);
      // k_h = tanh(A @ hW + bf); acc_h += w*k_h
      k_dh<<<64, 32, 0, stream>>>(A_bf, hWt, bfv, kcur_h, acc_h, wq[s]);
      // G = tanh(z_stage @ Wg + bg)   (2048 waves)
      k_gemm<4, 1><<<64 * 32, 32, 0, stream>>>(y_z, kprev_z, scale, HH, HH,
                                               Wgt_bf, bg, G, HH * DD, 32);
      // k_z = G . dX[idx]; acc_z += w*k_z
      k_dz<<<NN * HH / 256, 256, 0, stream>>>(G, dX + (size_t)idx * NN * DD,
                                              kcur_z, acc_z, wq[s]);
    }
    k_update<<<2 * NN * HH / 256, 256, 0, stream>>>(y, acc, dt / 6.0f);
  }

  // Phase 4: decode  out = z_final @ W_dec^T + b_dec
  k_gemm<4, 0><<<64 * 16, 32, 0, stream>>>(y_z, y_z, 0.0f, HH, HH, Wdec_bf,
                                           b_dec, (float*)d_out, NN, 16);
}
